// M2MicroTokenizer_48558900248781
// MI455X (gfx1250) — matrix operand order repaired and checked by reading, not compiled, not run on hardware
//
#include <hip/hip_runtime.h>
#include <cstdint>
#include <cstddef>

// ---------------------------------------------------------------------------
// CDNA5 (gfx1250, wave32) implementation.
// Pipeline:
//   0) one-time weight bf16-convert + swizzle into per-kstep slabs (ws)
//   1) density head: WMMA implicit-GEMM conv3x3 + GELU + 1x1, fused
//   2) bilinear importance sampling at analytic grid centers
//   3) per-batch top-8192 via bitonic sort of u64 keys in 256KB LDS
//   4) per-level gathered WMMA GEMM projecting ONLY the selected tokens
// B-slab fills use async global->LDS (b128) when the toolchain exposes it.
// ---------------------------------------------------------------------------

typedef __attribute__((ext_vector_type(16))) __bf16 v16bf;
typedef __attribute__((ext_vector_type(8)))  float  v8f;
typedef __attribute__((ext_vector_type(4)))  int    v4i;

union FragAB { unsigned int u[8]; v16bf v; };

__device__ __forceinline__ unsigned short f2bf(float f) {
  union { float f; unsigned int u; } x; x.f = f;
  unsigned int u = x.u;
  return (unsigned short)((u + 0x7FFFu + ((u >> 16) & 1u)) >> 16); // RNE
}

// output float offsets (concatenated tuple, return order)
#define SEL_TOK_OFF 0ull          // 8*8192*256
#define SEL_POS_OFF 16777216ull   // 8*8192*2
#define SEL_BOX_OFF 16908288ull   // 8*8192*4
#define TOPK_OFF    17170432ull   // 8*8192
#define DLOG_OFF    17235968ull   // 8*64*64

#define NTOK 21504
#define KSEL 8192
#define LDSTR   40   // LDS row stride in halves (80B: 16B aligned, bank-safe)
#define LDSTR32 20

// workspace byte offsets (all 16B aligned; total ~7.2 MB)
#define WS_DENSITY 0ull
#define WS_IMP     131072ull
#define WS_TOPKIDX 819200ull
#define WS_COUNTS  1081344ull
#define WS_LISTS   1081472ull
#define WS_W1BF    1867904ull    // 144*512*32 bf16 = 4718592 B
#define WS_PWBF0   6586496ull    // 8 steps *256*32 bf16
#define WS_PWBF1   6717568ull    // 16 steps
#define WS_PWBF2   6979712ull    // 32 steps

#if defined(__AMDGCN__) && \
    __has_builtin(__builtin_amdgcn_global_load_async_to_lds_b128) && \
    __has_builtin(__builtin_amdgcn_s_wait_asynccnt)
#define USE_ASYNC_LDS 1
#else
#define USE_ASYNC_LDS 0
#endif

// Load one contiguous 16KB bf16 slab (256 rows x 32 halves) into LDS with
// row stride LDSTR. 128 threads, 8 x b128 per thread.
__device__ __forceinline__ void load_b_slab(const unsigned short* __restrict__ slab,
                                            unsigned short* lds, int tid)
{
#if USE_ASYNC_LDS
  typedef __attribute__((address_space(1))) v4i gv4i;  // global
  typedef __attribute__((address_space(3))) v4i lv4i;  // LDS
  #pragma unroll
  for (int it = 0; it < 8; ++it) {
    int chunk = (it << 7) + tid;        // 0..1023 chunks of 16B
    int row = chunk >> 2, sub = chunk & 3;
    gv4i* gp = (gv4i*)((char*)slab + (size_t)chunk * 16);
    lv4i* lp = (lv4i*)((char*)lds + (size_t)(row * 80 + sub * 16));
    __builtin_amdgcn_global_load_async_to_lds_b128(gp, lp, 0, 0);
  }
#else
  const uint4* g = (const uint4*)slab;
  #pragma unroll
  for (int it = 0; it < 8; ++it) {
    int chunk = (it << 7) + tid;
    int row = chunk >> 2, sub = chunk & 3;
    *(uint4*)((char*)lds + row * 80 + sub * 16) = g[chunk];
  }
#endif
}

__device__ __forceinline__ void wait_async_zero() {
#if USE_ASYNC_LDS
  __builtin_amdgcn_s_wait_asynccnt(0);
#endif
}

__device__ __forceinline__ void decode_tok(int t, int& lvl, int& p, int& H, int& W) {
  if (t < 16384)      { lvl = 0; p = t;         H = 128; W = 128; }
  else if (t < 20480) { lvl = 1; p = t - 16384; H = 64;  W = 64;  }
  else                { lvl = 2; p = t - 20480; H = 32;  W = 32;  }
}

// ---------------------------------------------------------------------------
// Kernel 0a: dh1_w (512,512,3,3) f32 -> bf16 slabs [s=tap*16+cchunk][n][k]
// ---------------------------------------------------------------------------
__global__ void convert_w1_kernel(const float* __restrict__ w1,
                                  unsigned short* __restrict__ dst)
{
  int idx = blockIdx.x * blockDim.x + threadIdx.x;
  if (idx >= 144 * 512 * 32) return;
  int s = idx / (512 * 32);
  int r = idx % (512 * 32);
  int n = r >> 5, k = r & 31;
  int tap = s >> 4, c = ((s & 15) << 5) + k;
  dst[idx] = f2bf(w1[((size_t)n * 512 + c) * 9 + tap]);
}

// ---------------------------------------------------------------------------
// Kernel 0b: proj_w (256,C) f32 -> bf16 slabs [cstep][n][k]
// ---------------------------------------------------------------------------
__global__ void convert_pw_kernel(const float* __restrict__ w, int C,
                                  unsigned short* __restrict__ dst)
{
  int idx = blockIdx.x * blockDim.x + threadIdx.x;
  if (idx >= 256 * C) return;
  int cs = idx / (256 * 32);
  int r = idx % (256 * 32);
  int n = r >> 5, k = r & 31;
  dst[idx] = f2bf(w[(size_t)n * C + (cs << 5) + k]);
}

// ---------------------------------------------------------------------------
// Kernel 1: fused density head.
// M = 16 pixels/block, N = 512 (2 chunks of 256; 4 waves x 4 wmma-tiles),
// K = 9 taps * 512 cin = 4608 (144 steps of 32).
// Epilogue: GELU + dh2-weighted reduction over N via ds_add_f32.
// ---------------------------------------------------------------------------
__global__ __launch_bounds__(128) void density_head_kernel(
    const float* __restrict__ f8, const unsigned short* __restrict__ w1bf,
    const float* __restrict__ b1, const float* __restrict__ w2,
    const float* __restrict__ b2, float* __restrict__ out,
    float* __restrict__ density_ws)
{
  __shared__ unsigned short Alds[16 * LDSTR];
  __shared__ unsigned short Blds[256 * LDSTR];
  __shared__ float lds_logit[16];

  const int tid  = threadIdx.x;
  const int lane = tid & 31, wave = tid >> 5;
  const int half = lane >> 4, nlane = lane & 15;
  const int b    = blockIdx.y;
  const int tile = blockIdx.x;
  if (tid < 16) lds_logit[tid] = 0.f;

  const unsigned int* A32 = (const unsigned int*)Alds;
  const unsigned int* B32 = (const unsigned int*)Blds;
  const size_t f8b = (size_t)b * 512 * 4096;
  const int am = tid & 15, ak0 = (tid >> 4) << 2; // coalesced im2col mapping

  for (int nch = 0; nch < 2; ++nch) {
    v8f acc[4];
    for (int t = 0; t < 4; ++t)
      for (int r = 0; r < 8; ++r) acc[t][r] = 0.f;

    for (int s = 0; s < 144; ++s) {
      const int tap = s >> 4, c0 = (s & 15) << 5;
      const int dy = tap / 3 - 1, dx = tap % 3 - 1;
      __syncthreads();
      // B slab: pre-swizzled bf16 weights, contiguous 16KB (async -> LDS)
      load_b_slab(w1bf + ((size_t)s * 512 + (size_t)nch * 256) * 32, Blds, tid);
      { // A tile: im2col 16 pixels x 32 cin, zero-padded borders, f32->bf16
        int pix = tile * 16 + am;
        int y = (pix >> 6) + dy, x = (pix & 63) + dx;
        bool valid = (y >= 0 && y < 64 && x >= 0 && x < 64);
        size_t base = f8b + (size_t)(c0 + ak0) * 4096 + (size_t)(y * 64 + x);
        union { unsigned short h[4]; uint2 v; } pk;
        #pragma unroll
        for (int i = 0; i < 4; ++i)
          pk.h[i] = f2bf(valid ? f8[base + (size_t)i * 4096] : 0.f);
        *(uint2*)(&Alds[am * LDSTR + ak0]) = pk.v;
      }
      wait_async_zero();
      __syncthreads();
      FragAB af;
      #pragma unroll
      for (int v = 0; v < 8; ++v) { // ISA A layout: lane-half selects K group
        int k = (v < 4) ? (half * 8 + 2 * v) : (16 + half * 8 + 2 * (v - 4));
        af.u[v] = A32[nlane * LDSTR32 + (k >> 1)];
      }
      const int n0 = wave * 64;
      #pragma unroll
      for (int nt = 0; nt < 4; ++nt) {
        FragAB bf;
        int nl = n0 + nt * 16 + nlane;
        #pragma unroll
        for (int v = 0; v < 8; ++v)
          bf.u[v] = B32[nl * LDSTR32 + ((v < 4) ? (half * 4 + v) : (8 + half * 4 + (v - 4)))];
        acc[nt] = __builtin_amdgcn_wmma_f32_16x16x32_bf16(
            false, af.v, false, bf.v, (short)0, acc[nt], false, false);
      }
    }
    // epilogue: h = gelu(conv + b1); logit += h * w2
    #pragma unroll
    for (int nt = 0; nt < 4; ++nt) {
      int gn = nch * 256 + wave * 64 + nt * 16 + nlane;
      float bb = b1[gn], ww = w2[gn];
      #pragma unroll
      for (int r = 0; r < 8; ++r) {
        int m = half * 8 + r;
        float xv = acc[nt][r] + bb;
        float g = 0.5f * xv * (1.f + erff(xv * 0.70710678118654752f));
        atomicAdd(&lds_logit[m], g * ww);
      }
    }
  }
  __syncthreads();
  if (tid < 16) {
    int pix = tile * 16 + tid;
    float logit = lds_logit[tid] + b2[0];
    out[DLOG_OFF + (size_t)b * 4096 + pix] = logit;
    density_ws[(size_t)b * 4096 + pix] = 1.f / (1.f + expf(-logit));
  }
}

// ---------------------------------------------------------------------------
// Kernel 2: bilinear importance sampling at the 21504 analytic centers.
// ---------------------------------------------------------------------------
__global__ void importance_kernel(const float* __restrict__ density,
                                  float* __restrict__ imp)
{
  int id = blockIdx.x * blockDim.x + threadIdx.x;
  if (id >= 8 * NTOK) return;
  int b = id / NTOK, t = id % NTOK;
  int lvl, p, H, W; decode_tok(t, lvl, p, H, W);
  float cx = ((p % W) + 0.5f) / (float)W;
  float cy = ((p / W) + 0.5f) / (float)H;
  float gx = cx * 2.f - 1.f, gy = cy * 2.f - 1.f;
  float ix = ((gx + 1.f) * 64.f - 1.f) * 0.5f;
  float iy = ((gy + 1.f) * 64.f - 1.f) * 0.5f;
  float x0 = floorf(ix), y0 = floorf(iy);
  float wx = ix - x0, wy = iy - y0;
  int x0i = (int)x0, y0i = (int)y0;
  const float* D = density + (size_t)b * 4096;
  auto g = [&](int yy, int xx) -> float {
    bool valid = (xx >= 0 && xx < 64 && yy >= 0 && yy < 64);
    int yc = min(max(yy, 0), 63), xc = min(max(xx, 0), 63);
    float v = D[yc * 64 + xc];
    return valid ? v : 0.f;
  };
  float v = g(y0i, x0i) * (1.f - wx) * (1.f - wy)
          + g(y0i, x0i + 1) * wx * (1.f - wy)
          + g(y0i + 1, x0i) * (1.f - wx) * wy
          + g(y0i + 1, x0i + 1) * wx * wy;
  imp[(size_t)b * NTOK + t] = v;
}

// ---------------------------------------------------------------------------
// Kernel 3: per-batch top-8192 via bitonic sort of 32768 u64 keys in 256KB
// dynamic LDS (legal on CDNA5's 320KB/WG). Emits topk_vals, sel_pos,
// sel_boxes, and per-level compacted (row j | local pos) lists.
// ---------------------------------------------------------------------------
__global__ void topk_kernel(const float* __restrict__ imp, float* __restrict__ out,
                            unsigned* __restrict__ topkidx,
                            unsigned* __restrict__ counts,
                            unsigned* __restrict__ lists)
{
  extern __shared__ unsigned long long key[];
  __shared__ int cnt[3];
  const int tid = threadIdx.x, nthr = blockDim.x;
  const int b = blockIdx.x;
  const int N = 32768;
  if (tid < 3) cnt[tid] = 0;
  for (int i = tid; i < N; i += nthr) {
    unsigned long long kk = 0ull;
    if (i < NTOK) {
      union { float f; unsigned u; } x; x.f = imp[(size_t)b * NTOK + i];
      // importance = sigmoid(.) > 0 -> float bits order monotonically
      kk = ((unsigned long long)x.u << 32) |
           (unsigned long long)(0xFFFFFFFFu - (unsigned)i); // tie: lower idx first
    }
    key[i] = kk;
  }
  for (int k = 2; k <= N; k <<= 1)
    for (int j = k >> 1; j > 0; j >>= 1) {
      __syncthreads();
      for (int i = tid; i < N; i += nthr) {
        int ixj = i ^ j;
        if (ixj > i) {
          unsigned long long a = key[i], c = key[ixj];
          bool desc = ((i & k) == 0);
          bool sw = desc ? (a < c) : (a > c);
          if (sw) { key[i] = c; key[ixj] = a; }
        }
      }
    }
  __syncthreads();
  for (int j = tid; j < KSEL; j += nthr) {
    unsigned long long kk = key[j];
    union { unsigned u; float f; } vb; vb.u = (unsigned)(kk >> 32);
    unsigned t = 0xFFFFFFFFu - (unsigned)(kk & 0xFFFFFFFFull);
    out[TOPK_OFF + (size_t)b * KSEL + j] = vb.f;
    topkidx[(size_t)b * KSEL + j] = t;
    int lvl, p, H, W; decode_tok((int)t, lvl, p, H, W);
    float cx = ((p % W) + 0.5f) / (float)W;
    float cy = ((p / W) + 0.5f) / (float)H;
    size_t po = SEL_POS_OFF + ((size_t)b * KSEL + j) * 2;
    out[po] = cx; out[po + 1] = cy;
    float hx = 0.5f / (float)W, hy = 0.5f / (float)H;
    size_t bo = SEL_BOX_OFF + ((size_t)b * KSEL + j) * 4;
    out[bo]     = fminf(fmaxf(cx - hx, 0.f), 1.f);
    out[bo + 1] = fminf(fmaxf(cy - hy, 0.f), 1.f);
    out[bo + 2] = fminf(fmaxf(cx + hx, 0.f), 1.f);
    out[bo + 3] = fminf(fmaxf(cy + hy, 0.f), 1.f);
    int pos = atomicAdd(&cnt[lvl], 1);
    lists[((size_t)b * 3 + lvl) * KSEL + pos] =
        ((unsigned)j << 14) | (unsigned)p; // j:13b, p:14b
  }
  __syncthreads();
  if (tid < 3) counts[b * 3 + tid] = (unsigned)cnt[tid];
}

// ---------------------------------------------------------------------------
// Kernel 4: gathered projection GEMM for one pyramid level.
// M=16 gathered tokens/block, N=256, K=C (256/512/1024). Pre-swizzled bf16
// weights stream into LDS (async b128); scatter-store rows by selected j.
// ---------------------------------------------------------------------------
__global__ __launch_bounds__(128) void proj_gather_kernel(
    const float* __restrict__ feat, const unsigned short* __restrict__ wbf,
    const float* __restrict__ bias, int C, int HW, int lvl,
    const unsigned* __restrict__ lists, const unsigned* __restrict__ counts,
    float* __restrict__ out)
{
  const int b = blockIdx.y;
  const int count = (int)counts[b * 3 + lvl];
  const int tile = blockIdx.x;
  if (tile * 16 >= count) return;

  __shared__ unsigned short Alds[16 * LDSTR];
  __shared__ unsigned short Blds[256 * LDSTR];
  __shared__ int pm[16];
  __shared__ int jm[16];

  const int tid  = threadIdx.x;
  const int lane = tid & 31, wave = tid >> 5;
  const int half = lane >> 4, nlane = lane & 15;
  const int am = tid & 15, ak0 = (tid >> 4) << 2;

  if (tid < 16) {
    int row = tile * 16 + tid;
    if (row < count) {
      unsigned e = lists[((size_t)b * 3 + lvl) * KSEL + row];
      jm[tid] = (int)(e >> 14);
      pm[tid] = (int)(e & 0x3FFFu);
    } else { jm[tid] = -1; pm[tid] = 0; }
  }
  const unsigned int* A32 = (const unsigned int*)Alds;
  const unsigned int* B32 = (const unsigned int*)Blds;
  const size_t fb = (size_t)b * (size_t)C * (size_t)HW;

  v8f acc[4];
  for (int t = 0; t < 4; ++t)
    for (int r = 0; r < 8; ++r) acc[t][r] = 0.f;

  const int ksteps = C >> 5;
  for (int s = 0; s < ksteps; ++s) {
    const int c0 = s << 5;
    __syncthreads();
    load_b_slab(wbf + (size_t)s * 256 * 32, Blds, tid);
    { // A: gather 16 token columns x 32 channels, f32->bf16
      size_t base = fb + (size_t)(c0 + ak0) * HW + (size_t)pm[am];
      union { unsigned short h[4]; uint2 v; } pk;
      #pragma unroll
      for (int i = 0; i < 4; ++i)
        pk.h[i] = f2bf(feat[base + (size_t)i * HW]);
      *(uint2*)(&Alds[am * LDSTR + ak0]) = pk.v;
    }
    wait_async_zero();
    __syncthreads();
    FragAB af;
    #pragma unroll
    for (int v = 0; v < 8; ++v) {
      int k = (v < 4) ? (half * 8 + 2 * v) : (16 + half * 8 + 2 * (v - 4));
      af.u[v] = A32[nlane * LDSTR32 + (k >> 1)];
    }
    const int n0 = wave * 64;
    #pragma unroll
    for (int nt = 0; nt < 4; ++nt) {
      FragAB bf;
      int nl = n0 + nt * 16 + nlane;
      #pragma unroll
      for (int v = 0; v < 8; ++v)
        bf.u[v] = B32[nl * LDSTR32 + ((v < 4) ? (half * 4 + v) : (8 + half * 4 + (v - 4)))];
      acc[nt] = __builtin_amdgcn_wmma_f32_16x16x32_bf16(
          false, af.v, false, bf.v, (short)0, acc[nt], false, false);
    }
  }
  __syncthreads();
  #pragma unroll
  for (int nt = 0; nt < 4; ++nt) {
    int n = wave * 64 + nt * 16 + nlane;
    float bn = bias[n];
    #pragma unroll
    for (int r = 0; r < 8; ++r) {
      int m = half * 8 + r;
      int j = jm[m];
      if (j >= 0)
        out[SEL_TOK_OFF + ((size_t)b * KSEL + j) * 256 + n] = acc[nt][r] + bn;
    }
  }
}

// ---------------------------------------------------------------------------
extern "C" void kernel_launch(void* const* d_in, const int* in_sizes, int n_in,
                              void* d_out, int out_size, void* d_ws, size_t ws_size,
                              hipStream_t stream) {
  (void)in_sizes; (void)n_in; (void)out_size; (void)ws_size;
  const float* f4    = (const float*)d_in[0];
  const float* f8    = (const float*)d_in[1];
  const float* f16   = (const float*)d_in[2];
  const float* p4w   = (const float*)d_in[3];
  const float* p4b   = (const float*)d_in[4];
  const float* p8w   = (const float*)d_in[5];
  const float* p8b   = (const float*)d_in[6];
  const float* p16w  = (const float*)d_in[7];
  const float* p16b  = (const float*)d_in[8];
  const float* dh1w  = (const float*)d_in[9];
  const float* dh1b  = (const float*)d_in[10];
  const float* dh2w  = (const float*)d_in[11];
  const float* dh2b  = (const float*)d_in[12];
  float* out = (float*)d_out;

  char* ws = (char*)d_ws;
  float*          density    = (float*)(ws + WS_DENSITY);
  float*          importance = (float*)(ws + WS_IMP);
  unsigned*       topkidx    = (unsigned*)(ws + WS_TOPKIDX);
  unsigned*       counts     = (unsigned*)(ws + WS_COUNTS);
  unsigned*       lists      = (unsigned*)(ws + WS_LISTS);
  unsigned short* w1bf       = (unsigned short*)(ws + WS_W1BF);
  unsigned short* pwbf0      = (unsigned short*)(ws + WS_PWBF0);
  unsigned short* pwbf1      = (unsigned short*)(ws + WS_PWBF1);
  unsigned short* pwbf2      = (unsigned short*)(ws + WS_PWBF2);

  // 0) one-time weight convert + swizzle (bf16 slabs)
  convert_w1_kernel<<<9216, 256, 0, stream>>>(dh1w, w1bf);
  convert_pw_kernel<<<256, 256, 0, stream>>>(p4w, 256, pwbf0);
  convert_pw_kernel<<<512, 256, 0, stream>>>(p8w, 512, pwbf1);
  convert_pw_kernel<<<1024, 256, 0, stream>>>(p16w, 1024, pwbf2);

  // 1..4) pipeline
  density_head_kernel<<<dim3(256, 8), 128, 0, stream>>>(
      f8, w1bf, dh1b, dh2w, dh2b, out, density);
  importance_kernel<<<672, 256, 0, stream>>>(density, importance);
  topk_kernel<<<8, 512, 32768 * sizeof(unsigned long long), stream>>>(
      importance, out, topkidx, counts, lists);
  proj_gather_kernel<<<dim3(512, 8), 128, 0, stream>>>(
      f4, pwbf0, p4b, 256, 16384, 0, lists, counts, out);
  proj_gather_kernel<<<dim3(512, 8), 128, 0, stream>>>(
      f8, pwbf1, p8b, 512, 4096, 1, lists, counts, out);
  proj_gather_kernel<<<dim3(512, 8), 128, 0, stream>>>(
      f16, pwbf2, p16b, 1024, 1024, 2, lists, counts, out);
}
